// PolicyValueNetV4_50182397886690
// MI455X (gfx1250) — matrix operand order repaired and checked by reading, not compiled, not run on hardware
//
#include <hip/hip_runtime.h>
#include <hip/hip_bf16.h>
#include <math.h>

typedef __bf16 bf16;
typedef __attribute__((ext_vector_type(16))) __bf16 v16bf;
typedef __attribute__((ext_vector_type(8)))  __bf16 v8bf;
typedef __attribute__((ext_vector_type(8)))  float  v8f;

#define HH 8
#define DDIM 32
#define CCH 256
#define KNODES 88
#define ACTN 4644

// ---------------- utility kernels ----------------

__global__ void fill_kernel(float* __restrict__ p, float v, int n) {
  int i = blockIdx.x * blockDim.x + threadIdx.x;
  if (i < n) p[i] = v;
}

// Pack fp32 weight B[K,Ncol] into WMMA B-fragment layout (bf16), zero-padded.
// Bp[((ntile*ktiles + kt)*32 + lane)*16 + e] = B[(kt*32 + ((lane&16)>>1) + (e&7) + ((e&8)<<1))*Ncol + ntile*16 + (lane&15)]
__global__ void pack_b_kernel(const float* __restrict__ B, bf16* __restrict__ Bp,
                              int K, int Ncol, int ntiles_pad) {
  int idx = blockIdx.x * blockDim.x + threadIdx.x;
  int ktiles = K >> 5;
  int total = ntiles_pad * ktiles * 512;
  if (idx >= total) return;
  int e = idx & 15;
  int lane = (idx >> 4) & 31;
  int kt = (idx >> 9) % ktiles;
  int ntile = (idx >> 9) / ktiles;
  int col = (ntile << 4) + (lane & 15);
  int k = (kt << 5) + ((lane & 16) >> 1) + (e & 7) + ((e & 8) << 1);
  float v = (col < Ncol) ? B[(size_t)k * Ncol + col] : 0.f;
  Bp[idx] = (bf16)v;
}

// h = relu(x @ W_emb + b) + pos_emb[n % 88];  also bf16 shadow
__global__ void embed_kernel(const float* __restrict__ x, const float* __restrict__ W,
                             const float* __restrict__ b, const float* __restrict__ pos,
                             float* __restrict__ h, bf16* __restrict__ hbf, int n_nodes) {
  int i = blockIdx.x * blockDim.x + threadIdx.x;
  if (i >= n_nodes * CCH) return;
  int n = i >> 8, c = i & 255;
  float acc = b[c];
  const float* xr = x + n * 16;
#pragma unroll
  for (int j = 0; j < 16; ++j) acc += xr[j] * W[j * CCH + c];
  acc = acc > 0.f ? acc : 0.f;
  acc += pos[(n % KNODES) * CCH + c];
  h[i] = acc;
  hbf[i] = (bf16)acc;
}

// ---------------- WMMA GEMM ----------------
// C[M,Ncol] = A[M,K](bf16, row-major) x Bpacked (+bias)(+relu).
// One wave computes a 16x64 strip: 4 N-tiles share one A fragment.
__global__ __launch_bounds__(256) void gemm_bf16_kernel(
    const bf16* __restrict__ A, const bf16* __restrict__ Bp,
    const float* __restrict__ bias, float* __restrict__ C,
    int M, int K, int Ncol, int nquads, int relu_flag) {
  int wave = (blockIdx.x * blockDim.x + threadIdx.x) >> 5;
  int lane = threadIdx.x & 31;
  int mtiles = M >> 4;
  if (wave >= mtiles * nquads) return;
  int mtile = wave / nquads, nquad = wave % nquads;
  int row = (mtile << 4) + (lane & 15);
  bool hi = lane >= 16;
  int ktiles = K >> 5;

  v8f acc[4];
#pragma unroll
  for (int q = 0; q < 4; ++q) acc[q] = (v8f){};

  const bf16* arow = A + (size_t)row * K + (hi ? 8 : 0);
  const size_t tileStride = (size_t)ktiles * 512;  // elements between consecutive ntiles
  const bf16* bquad = Bp + (size_t)(nquad * 4) * tileStride + lane * 16;

  for (int kt = 0; kt < ktiles; ++kt) {
    // A fragment: two contiguous 16B chunks per lane
    v8bf a0 = *(const v8bf*)(arow + (kt << 5));
    v8bf a1 = *(const v8bf*)(arow + (kt << 5) + 16);
    v16bf af = __builtin_shufflevector(a0, a1, 0, 1, 2, 3, 4, 5, 6, 7,
                                       8, 9, 10, 11, 12, 13, 14, 15);
    const bf16* bk = bquad + (size_t)kt * 512;
#pragma unroll
    for (int q = 0; q < 4; ++q) {
      v16bf bfv = *(const v16bf*)(bk + (size_t)q * tileStride);
      acc[q] = __builtin_amdgcn_wmma_f32_16x16x32_bf16(false, af, false, bfv, (short)0,
                                                       acc[q], false, false);
    }
  }

#pragma unroll
  for (int q = 0; q < 4; ++q) {
    int ntile = nquad * 4 + q;
    int col = (ntile << 4) + (lane & 15);
    if (col >= Ncol) continue;
    float bc = (bias != nullptr) ? bias[col] : 0.f;
#pragma unroll
    for (int i = 0; i < 8; ++i) {
      int r = (mtile << 4) + i + (hi ? 8 : 0);
      float v = acc[q][i] + bc;
      if (relu_flag) v = v > 0.f ? v : 0.f;
      C[(size_t)r * Ncol + col] = v;
    }
  }
}

// ---------------- edge (GATv2) kernels ----------------

__device__ inline void atomicMaxF(float* addr, float val) {
  int* ia = (int*)addr;
  int old = __float_as_int(*addr);
  while (__int_as_float(old) < val) {
    int assumed = old;
    old = atomicCAS(ia, assumed, __float_as_int(val));
    if (old == assumed) break;
  }
}

__global__ void edge_passA_kernel(const int* __restrict__ src, const int* __restrict__ dst,
                                  const float* __restrict__ xl, const float* __restrict__ xr,
                                  const float* __restrict__ att, float* __restrict__ esc,
                                  float* __restrict__ mmax, int E) {
  int t = blockIdx.x * blockDim.x + threadIdx.x;
  if (t >= E * HH) return;
  int e = t >> 3, h = t & 7;
  int s = src[e], d = dst[e];
  const float* pl = xl + (size_t)s * CCH + h * DDIM;
  const float* pr = xr + (size_t)d * CCH + h * DDIM;
  const float* pa = att + h * DDIM;
  float acc = 0.f;
#pragma unroll
  for (int k = 0; k < DDIM; ++k) {
    float z = pl[k] + pr[k];
    z = z > 0.f ? z : 0.2f * z;
    acc += z * pa[k];
  }
  esc[t] = acc;
  atomicMaxF(mmax + d * HH + h, acc);
}

__global__ void edge_passB_kernel(const int* __restrict__ dst, float* __restrict__ esc,
                                  const float* __restrict__ mmax, float* __restrict__ dsum,
                                  int E) {
  int t = blockIdx.x * blockDim.x + threadIdx.x;
  if (t >= E * HH) return;
  int e = t >> 3, h = t & 7;
  int d = dst[e];
  float a = __expf(esc[t] - mmax[d * HH + h]);
  esc[t] = a;
  atomicAdd(dsum + d * HH + h, a);
}

__global__ void edge_passC_kernel(const int* __restrict__ src, const int* __restrict__ dst,
                                  const float* __restrict__ esc, const float* __restrict__ dsum,
                                  const float* __restrict__ xl, float* __restrict__ outacc,
                                  int E) {
  int t = blockIdx.x * blockDim.x + threadIdx.x;
  if (t >= E * HH) return;
  int e = t >> 3, h = t & 7;
  int s = src[e], d = dst[e];
  float w = esc[t] / dsum[d * HH + h];
  const float* pl = xl + (size_t)s * CCH + h * DDIM;
  float* po = outacc + (size_t)d * CCH + h * DDIM;
#pragma unroll
  for (int k = 0; k < DDIM; ++k) atomicAdd(po + k, w * pl[k]);
}

// h = LayerNorm(h + relu(outacc + bias)) * g + b ; also bf16 shadow
__global__ void node_update_kernel(float* __restrict__ h, bf16* __restrict__ hbf,
                                   const float* __restrict__ outacc,
                                   const float* __restrict__ bias,
                                   const float* __restrict__ lng,
                                   const float* __restrict__ lnb) {
  __shared__ float red[256];
  int n = blockIdx.x, c = threadIdx.x;
  size_t idx = (size_t)n * CCH + c;
  float o = outacc[idx] + bias[c];
  o = o > 0.f ? o : 0.f;
  float y = h[idx] + o;
  red[c] = y;
  __syncthreads();
  for (int s = 128; s > 0; s >>= 1) { if (c < s) red[c] += red[c + s]; __syncthreads(); }
  float mu = red[0] * (1.f / 256.f);
  __syncthreads();
  float dv = y - mu;
  red[c] = dv * dv;
  __syncthreads();
  for (int s = 128; s > 0; s >>= 1) { if (c < s) red[c] += red[c + s]; __syncthreads(); }
  float var = red[0] * (1.f / 256.f);
  float out = dv * rsqrtf(var + 1e-5f) * lng[c] + lnb[c];
  h[idx] = out;
  hbf[idx] = (bf16)out;
}

// gate[n] = g1[n,:] . gW2 + gb2
__global__ void gate_dot_kernel(const float* __restrict__ g1, const float* __restrict__ gW2,
                                const float* __restrict__ gb2, float* __restrict__ gate,
                                int n_nodes) {
  int n = blockIdx.x * blockDim.x + threadIdx.x;
  if (n >= n_nodes) return;
  float s = gb2[0];
  const float* p = g1 + (size_t)n * 128;
#pragma unroll 4
  for (int j = 0; j < 128; ++j) s += p[j] * gW2[j];
  gate[n] = s;
}

// per-graph gated-softmax pooling; nodes of graph g are rows g*88 .. g*88+87
__global__ void pool_kernel(const float* __restrict__ h, const float* __restrict__ gate,
                            float* __restrict__ emb, bf16* __restrict__ embbf) {
  __shared__ float wsh[KNODES];
  __shared__ float red[128];
  int g = blockIdx.x, t = threadIdx.x;
  const float* gg = gate + g * KNODES;
  if (t < 128) {
    float m = -3.402823466e38f;
    for (int i = t; i < KNODES; i += 128) m = fmaxf(m, gg[i]);
    red[t] = m;
  }
  __syncthreads();
  for (int s = 64; s > 0; s >>= 1) { if (t < s) red[t] = fmaxf(red[t], red[t + s]); __syncthreads(); }
  float m = red[0];
  __syncthreads();
  if (t < 128) {
    float acc = 0.f;
    for (int i = t; i < KNODES; i += 128) { float a = __expf(gg[i] - m); wsh[i] = a; acc += a; }
    red[t] = acc;
  }
  __syncthreads();
  for (int s = 64; s > 0; s >>= 1) { if (t < s) red[t] += red[t + s]; __syncthreads(); }
  float denom = red[0];
  __syncthreads();
  int c = t;
  const float* hb = h + (size_t)g * KNODES * CCH;
  float acc = 0.f;
  for (int i = 0; i < KNODES; ++i) acc += wsh[i] * hb[(size_t)i * CCH + c];
  acc /= denom;
  emb[(size_t)g * CCH + c] = acc;
  embbf[(size_t)g * CCH + c] = (bf16)acc;
}

// relu(LayerNorm(in)) -> bf16
__global__ void ln_relu_bf16_kernel(const float* __restrict__ in, const float* __restrict__ g,
                                    const float* __restrict__ b, bf16* __restrict__ out) {
  __shared__ float red[256];
  int n = blockIdx.x, c = threadIdx.x;
  size_t idx = (size_t)n * CCH + c;
  float y = in[idx];
  red[c] = y;
  __syncthreads();
  for (int s = 128; s > 0; s >>= 1) { if (c < s) red[c] += red[c + s]; __syncthreads(); }
  float mu = red[0] * (1.f / 256.f);
  __syncthreads();
  float dv = y - mu;
  red[c] = dv * dv;
  __syncthreads();
  for (int s = 128; s > 0; s >>= 1) { if (c < s) red[c] += red[c + s]; __syncthreads(); }
  float var = red[0] * (1.f / 256.f);
  float v = dv * rsqrtf(var + 1e-5f) * g[c] + b[c];
  v = v > 0.f ? v : 0.f;
  out[idx] = (bf16)v;
}

// value[g] = tanh(vh[g,:] . vW2 + vb2)
__global__ void value_kernel(const float* __restrict__ vh, const float* __restrict__ vW2,
                             const float* __restrict__ vb2, float* __restrict__ out, int Gn) {
  int g = blockIdx.x * blockDim.x + threadIdx.x;
  if (g >= Gn) return;
  float s = vb2[0];
  const float* p = vh + (size_t)g * CCH;
#pragma unroll 4
  for (int j = 0; j < CCH; ++j) s += p[j] * vW2[j];
  out[g] = tanhf(s);
}

// ---------------- host launch ----------------

static inline int ntiles_pad_of(int Ncol) {
  int nt = (Ncol + 15) >> 4;
  return (nt + 3) & ~3;
}

extern "C" void kernel_launch(void* const* d_in, const int* in_sizes, int n_in,
                              void* d_out, int out_size, void* d_ws, size_t ws_size,
                              hipStream_t stream) {
  const float* x        = (const float*)d_in[0];
  const int*   ei       = (const int*)d_in[1];
  const float* W_emb    = (const float*)d_in[4];
  const float* b_emb    = (const float*)d_in[5];
  const float* pos_emb  = (const float*)d_in[6];
  const float* gat_Wsrc = (const float*)d_in[7];
  const float* gat_Wdst = (const float*)d_in[8];
  const float* gat_att  = (const float*)d_in[9];
  const float* gat_bias = (const float*)d_in[10];
  const float* ln_g     = (const float*)d_in[11];
  const float* ln_b     = (const float*)d_in[12];
  const float* gW1      = (const float*)d_in[13];
  const float* gb1      = (const float*)d_in[14];
  const float* gW2      = (const float*)d_in[15];
  const float* gb2      = (const float*)d_in[16];
  const float* pW1      = (const float*)d_in[17];
  const float* pb1      = (const float*)d_in[18];
  const float* p_ln_g   = (const float*)d_in[19];
  const float* p_ln_b   = (const float*)d_in[20];
  const float* pW2      = (const float*)d_in[21];
  const float* pb2      = (const float*)d_in[22];
  const float* vW1      = (const float*)d_in[23];
  const float* vb1      = (const float*)d_in[24];
  const float* vW2      = (const float*)d_in[25];
  const float* vb2      = (const float*)d_in[26];

  const int Nn = in_sizes[0] / 16;     // 22528 nodes
  const int E  = in_sizes[1] / 2;      // edges (incl. self loops)
  const int Gn = Nn / KNODES;          // 256 graphs
  const int* srcI = ei;
  const int* dstI = ei + E;

  // workspace carve-out
  char* wsp = (char*)d_ws;
  auto alloc = [&](size_t bytes) -> void* {
    void* p = (void*)wsp;
    wsp += (bytes + 255) & ~(size_t)255;
    return p;
  };
  const int ntp256 = ntiles_pad_of(CCH);    // 16
  const int ntp128 = ntiles_pad_of(128);    // 8
  const int ntpACT = ntiles_pad_of(ACTN);   // 292

  float* h      = (float*)alloc((size_t)Nn * CCH * 4);
  bf16*  hbf    = (bf16*)alloc((size_t)Nn * CCH * 2);
  float* xl     = (float*)alloc((size_t)Nn * CCH * 4);
  float* xr     = (float*)alloc((size_t)Nn * CCH * 4);
  float* outacc = (float*)alloc((size_t)Nn * CCH * 4);
  float* esc    = (float*)alloc((size_t)E * HH * 4);
  float* mmax   = (float*)alloc((size_t)Nn * HH * 4);
  float* dsum   = (float*)alloc((size_t)Nn * HH * 4);
  float* g1     = (float*)alloc((size_t)Nn * 128 * 4);
  float* gate   = (float*)alloc((size_t)Nn * 4);
  float* emb    = (float*)alloc((size_t)Gn * CCH * 4);
  bf16*  embbf  = (bf16*)alloc((size_t)Gn * CCH * 2);
  float* hid    = (float*)alloc((size_t)Gn * CCH * 4);
  bf16*  hidbf  = (bf16*)alloc((size_t)Gn * CCH * 2);
  float* vh     = (float*)alloc((size_t)Gn * CCH * 4);
  bf16*  wsrcp  = (bf16*)alloc((size_t)4 * CCH * ntp256 * 16 * 2);
  bf16*  wdstp  = (bf16*)alloc((size_t)4 * CCH * ntp256 * 16 * 2);
  bf16*  gW1p   = (bf16*)alloc((size_t)CCH * ntp128 * 16 * 2);
  bf16*  pW1p   = (bf16*)alloc((size_t)CCH * ntp256 * 16 * 2);
  bf16*  pW2p   = (bf16*)alloc((size_t)CCH * ntpACT * 16 * 2);
  bf16*  vW1p   = (bf16*)alloc((size_t)CCH * ntp256 * 16 * 2);

  auto pack = [&](const float* B, bf16* Bp, int K, int Ncol) {
    int ntp = ntiles_pad_of(Ncol);
    int total = ntp * (K >> 5) * 512;
    pack_b_kernel<<<(total + 255) / 256, 256, 0, stream>>>(B, Bp, K, Ncol, ntp);
  };
  auto fill = [&](float* p, float v, int n) {
    fill_kernel<<<(n + 255) / 256, 256, 0, stream>>>(p, v, n);
  };
  auto gemm = [&](const bf16* A, const bf16* Bp, const float* bias, float* C,
                  int M, int K, int Ncol, int relu) {
    int nquads = ntiles_pad_of(Ncol) >> 2;
    int tiles = (M >> 4) * nquads;
    int blocks = (tiles + 7) / 8;  // 8 waves/block
    gemm_bf16_kernel<<<blocks, 256, 0, stream>>>(A, Bp, bias, C, M, K, Ncol, nquads, relu);
  };

  // weight packing (deterministic each call)
  const size_t wmatP = (size_t)CCH * ntp256 * 16;  // packed elements per 256x256 matrix
  for (int l = 0; l < 4; ++l) {
    pack(gat_Wsrc + (size_t)l * CCH * CCH, wsrcp + (size_t)l * wmatP, CCH, CCH);
    pack(gat_Wdst + (size_t)l * CCH * CCH, wdstp + (size_t)l * wmatP, CCH, CCH);
  }
  pack(gW1, gW1p, CCH, 128);
  pack(pW1, pW1p, CCH, CCH);
  pack(pW2, pW2p, CCH, ACTN);
  pack(vW1, vW1p, CCH, CCH);

  // embedding
  embed_kernel<<<((size_t)Nn * CCH + 255) / 256, 256, 0, stream>>>(
      x, W_emb, b_emb, pos_emb, h, hbf, Nn);

  const int eThreads = E * HH;
  const int eBlocks = (eThreads + 255) / 256;

  for (int l = 0; l < 4; ++l) {
    gemm(hbf, wsrcp + (size_t)l * wmatP, nullptr, xl, Nn, CCH, CCH, 0);
    gemm(hbf, wdstp + (size_t)l * wmatP, nullptr, xr, Nn, CCH, CCH, 0);
    fill(mmax, -3.402823466e38f, Nn * HH);
    fill(dsum, 0.f, Nn * HH);
    fill(outacc, 0.f, Nn * CCH);
    edge_passA_kernel<<<eBlocks, 256, 0, stream>>>(srcI, dstI, xl, xr,
                                                   gat_att + l * HH * DDIM, esc, mmax, E);
    edge_passB_kernel<<<eBlocks, 256, 0, stream>>>(dstI, esc, mmax, dsum, E);
    edge_passC_kernel<<<eBlocks, 256, 0, stream>>>(srcI, dstI, esc, dsum, xl, outacc, E);
    node_update_kernel<<<Nn, 256, 0, stream>>>(h, hbf, outacc, gat_bias + l * CCH,
                                               ln_g + l * CCH, ln_b + l * CCH);
  }

  // gated pooling
  gemm(hbf, gW1p, gb1, g1, Nn, CCH, 128, 1);
  gate_dot_kernel<<<(Nn + 255) / 256, 256, 0, stream>>>(g1, gW2, gb2, gate, Nn);
  pool_kernel<<<Gn, 256, 0, stream>>>(h, gate, emb, embbf);

  // policy head
  float* policy_out = (float*)d_out;
  gemm(embbf, pW1p, pb1, hid, Gn, CCH, CCH, 0);
  ln_relu_bf16_kernel<<<Gn, 256, 0, stream>>>(hid, p_ln_g, p_ln_b, hidbf);
  gemm(hidbf, pW2p, pb2, policy_out, Gn, CCH, ACTN, 0);

  // value head
  gemm(embbf, vW1p, vb1, vh, Gn, CCH, CCH, 1);
  value_kernel<<<(Gn + 255) / 256, 256, 0, stream>>>(vh, vW2, vb2,
                                                     policy_out + (size_t)Gn * ACTN, Gn);
}
